// ScaledDotProductAttention_13314398618003
// MI455X (gfx1250) — matrix operand order repaired and checked by reading, not compiled, not run on hardware
//
#include <hip/hip_runtime.h>

// Flash attention for B=4,H=16,S=2048,D=64 on gfx1250 (wave32, WMMA).
// f16 WMMA with f32 accumulation; online softmax with deferred l-reduction;
// permuted key index inside each block so P spills are vector ds_store_b64.

typedef _Float16 half4  __attribute__((ext_vector_type(4)));
typedef _Float16 half8  __attribute__((ext_vector_type(8)));
typedef _Float16 half16 __attribute__((ext_vector_type(16)));
typedef float    v8f    __attribute__((ext_vector_type(8)));

#define S_LEN  2048
#define D_DIM  64
#define BN     64      // keys per block
#define BM     128     // query rows per workgroup (8 waves * 16)
#define NWAVES 8
#define PITCH  72      // halves per LDS row (64 + 8 pad, 144B, 16B-aligned)

__device__ __forceinline__ half16 cat16(half8 a, half8 b) {
  return __builtin_shufflevector(a, b, 0,1,2,3,4,5,6,7,8,9,10,11,12,13,14,15);
}

__global__ __launch_bounds__(256, 1)
void fa_fwd_kernel(const float* __restrict__ q, const float* __restrict__ k,
                   const float* __restrict__ v, float* __restrict__ out) {
  __shared__ alignas(16) _Float16 Kb[BN][PITCH];           // [t_local][d]       (transposed K)
  __shared__ alignas(16) _Float16 Vb[D_DIM][PITCH];        // [d][perm(t_local)] (transposed V)
  __shared__ alignas(16) _Float16 Pb[NWAVES][16][PITCH];   // per-wave probs [m][perm(t_local)]

  const int bh = blockIdx.y;
  const int q0 = blockIdx.x * BM;
  const float* Q = q + (size_t)bh * S_LEN * D_DIM;
  const float* K = k + (size_t)bh * D_DIM * S_LEN;   // k is [d][t]
  const float* V = v + (size_t)bh * S_LEN * D_DIM;   // v is [t][d]
  float*       O = out + (size_t)bh * S_LEN * D_DIM;

  const int tid  = threadIdx.x;
  const int wave = tid >> 5;
  const int lane = tid & 31;
  const int hi   = lane >> 4;   // which half of the wave
  const int ln   = lane & 15;

  // ---- Q tile A-operand (lane = row m, elements walk K per ISA layout) ----
  const float* qrow = Q + (size_t)(q0 + wave * 16 + ln) * D_DIM;
  half16 qa[2];
#pragma unroll
  for (int s = 0; s < 2; ++s) {
    const float4* f0 = (const float4*)(qrow + 32 * s + hi * 8);
    const float4* f1 = (const float4*)(qrow + 32 * s + hi * 8 + 16);
    float4 a = f0[0], b = f0[1], c = f1[0], d = f1[1];
    half16 t;
    t[0]=(_Float16)a.x;  t[1]=(_Float16)a.y;  t[2]=(_Float16)a.z;  t[3]=(_Float16)a.w;
    t[4]=(_Float16)b.x;  t[5]=(_Float16)b.y;  t[6]=(_Float16)b.z;  t[7]=(_Float16)b.w;
    t[8]=(_Float16)c.x;  t[9]=(_Float16)c.y;  t[10]=(_Float16)c.z; t[11]=(_Float16)c.w;
    t[12]=(_Float16)d.x; t[13]=(_Float16)d.y; t[14]=(_Float16)d.z; t[15]=(_Float16)d.w;
    qa[s] = t;
  }

  v8f o_acc[4];
#pragma unroll
  for (int nt = 0; nt < 4; ++nt)
#pragma unroll
    for (int r = 0; r < 8; ++r) o_acc[nt][r] = 0.f;

  float m_i[8], l_part[8];   // l_part: per-lane partial row sums (reduced at end)
#pragma unroll
  for (int r = 0; r < 8; ++r) { m_i[r] = -__builtin_inff(); l_part[r] = 0.f; }

  const float LOG2E = 1.44269504088896f;

  for (int kb = 0; kb < S_LEN / BN; ++kb) {
    const int t0 = kb * BN;

    // ---- cooperative staging: K block ([d][t] -> Kb[t][d], f16, b64 stores) ----
    {
      const int t    = tid & 63;
      const int dgrp = tid >> 6;        // 0..3
#pragma unroll
      for (int kk = 0; kk < 4; ++kk) {
        const int d = dgrp * 4 + kk * 16;
        half4 pk;
#pragma unroll
        for (int j = 0; j < 4; ++j)
          pk[j] = (_Float16)K[(size_t)(d + j) * S_LEN + (t0 + t)];
        *(half4*)&Kb[t][d] = pk;
      }
      // ---- V block ([t][d] -> Vb[d][perm(t)], f16, b64 stores) ----
      // perm(t) = (t & 15)*4 + (t >> 4): keys (tl, tl+16, tl+32, tl+48) contiguous.
      const int d  = tid & 63;
      const int tg = tid >> 6;          // 0..3
#pragma unroll
      for (int kk = 0; kk < 4; ++kk) {
        const int tl = tg + kk * 4;     // 0..15
        half4 pv;
#pragma unroll
        for (int j = 0; j < 4; ++j)
          pv[j] = (_Float16)V[(size_t)(t0 + tl + 16 * j) * D_DIM + d];
        *(half4*)&Vb[d][tl * 4] = pv;
      }
      // ---- prefetch next K/V block (global_prefetch_b8) ----
      if (kb + 1 < S_LEN / BN) {
        const int tn = t0 + BN;
        __builtin_prefetch(K + (size_t)(tid >> 2) * S_LEN + tn + (tid & 3) * 16, 0, 0);
        __builtin_prefetch(V + (size_t)(tn + (tid >> 2)) * D_DIM + (tid & 3) * 16, 0, 0);
      }
    }
    __syncthreads();

    // ---- S_tile = Q (16x64) x K^T (64x64): 4 N-tiles x 2 K-steps ----
    v8f st[4];
#pragma unroll
    for (int nt = 0; nt < 4; ++nt) {
      v8f acc;
#pragma unroll
      for (int r = 0; r < 8; ++r) acc[r] = 0.f;
#pragma unroll
      for (int s = 0; s < 2; ++s) {
        const half8* bp = (const half8*)&Kb[nt * 16 + ln][32 * s + hi * 8];
        half16 bk = cat16(bp[0], bp[2]);
        acc = __builtin_amdgcn_wmma_f32_16x16x32_f16(false, qa[s], false, bk,
                                                     (short)0, acc, false, false);
      }
      st[nt] = acc;
    }

    // ---- online softmax; lane handles rows m = r + 8*hi ----
#pragma unroll
    for (int r = 0; r < 8; ++r) {
      float mloc = fmaxf(fmaxf(st[0][r], st[1][r]), fmaxf(st[2][r], st[3][r]));
#pragma unroll
      for (int off = 8; off > 0; off >>= 1)
        mloc = fmaxf(mloc, __shfl_xor(mloc, off, 16));
      const float mnew  = fmaxf(m_i[r], mloc);
      const float scale = exp2f((m_i[r] - mnew) * LOG2E);
      float lsum = 0.f;
      half4 pk;
#pragma unroll
      for (int nt = 0; nt < 4; ++nt) {
        const float p = exp2f((st[nt][r] - mnew) * LOG2E);
        lsum += p;
        pk[nt] = (_Float16)p;            // perm column = ln*4 + nt
      }
      *(half4*)&Pb[wave][r + 8 * hi][ln * 4] = pk;   // one ds_store_b64
      l_part[r] = l_part[r] * scale + lsum;          // per-lane partial, no reduce
      m_i[r] = mnew;
#pragma unroll
      for (int nt = 0; nt < 4; ++nt) o_acc[nt][r] *= scale;
    }

    // ---- O += P (16xBN) x V (BNx64); LDS in-order per wave covers P RAW ----
#pragma unroll
    for (int s = 0; s < 2; ++s) {
      const half8* ap0 = (const half8*)&Pb[wave][ln][32 * s + hi * 8];
      const half8* ap1 = (const half8*)&Pb[wave][ln][32 * s + hi * 8 + 16];
      half16 pa = cat16(ap0[0], ap1[0]);
#pragma unroll
      for (int nt = 0; nt < 4; ++nt) {
        const half8* bp = (const half8*)&Vb[nt * 16 + ln][32 * s + hi * 8];
        half16 bv = cat16(bp[0], bp[2]);
        o_acc[nt] = __builtin_amdgcn_wmma_f32_16x16x32_f16(false, pa, false, bv,
                                                           (short)0, o_acc[nt], false, false);
      }
    }
    __syncthreads();   // protect Kb/Vb before next staging pass
  }

  // ---- epilogue: reduce row sums across the 16-lane half, normalize, store ----
  float* orow = O + (size_t)(q0 + wave * 16) * D_DIM;
#pragma unroll
  for (int r = 0; r < 8; ++r) {
    float l = l_part[r];
#pragma unroll
    for (int off = 8; off > 0; off >>= 1)
      l += __shfl_xor(l, off, 16);
    const float inv = 1.f / l;
    const int m = r + 8 * hi;
#pragma unroll
    for (int nt = 0; nt < 4; ++nt)
      orow[(size_t)m * D_DIM + nt * 16 + ln] = o_acc[nt][r] * inv;
  }
}

extern "C" void kernel_launch(void* const* d_in, const int* in_sizes, int n_in,
                              void* d_out, int out_size, void* d_ws, size_t ws_size,
                              hipStream_t stream) {
  const float* q = (const float*)d_in[0];
  const float* k = (const float*)d_in[1];
  const float* v = (const float*)d_in[2];
  float* out = (float*)d_out;
  dim3 grid(S_LEN / BM, 4 * 16);   // (S blocks, B*H)
  fa_fwd_kernel<<<grid, 256, 0, stream>>>(q, k, v, out);
}
